// MoE_50216757625284
// MI455X (gfx1250) — matrix operand order repaired and checked by reading, not compiled, not run on hardware
//
#include <hip/hip_runtime.h>
#include <hip/hip_bf16.h>
#include <math.h>

// ---------------------------------------------------------------------------
// MoE top-2 (E=16, D=O=512, H=128) for gfx1250 (CDNA5, wave32, WMMA bf16)
// ---------------------------------------------------------------------------

typedef __bf16 bf16;
typedef bf16  v16bf __attribute__((ext_vector_type(16)));
typedef bf16  v8bf  __attribute__((ext_vector_type(8)));
typedef float v8f   __attribute__((ext_vector_type(8)));
typedef int   v4i   __attribute__((vector_size(4 * sizeof(int))));

#define DIM_D 512
#define DIM_H 128
#define DIM_E 16
#define DIM_O 512

#define AS1 __attribute__((address_space(1)))
#define AS3 __attribute__((address_space(3)))

__device__ inline v8f v8f_zero() {
    v8f z;
#pragma unroll
    for (int i = 0; i < 8; ++i) z[i] = 0.0f;
    return z;
}

// ---- CDNA5 async global->LDS copy (ASYNCcnt-tracked) ----------------------
__device__ inline void async_copy_b128(void* lds, const void* g) {
#if __has_builtin(__builtin_amdgcn_global_load_async_to_lds_b128)
    __builtin_amdgcn_global_load_async_to_lds_b128(
        (AS1 v4i*)g, (AS3 v4i*)lds, 0, 0);
#else
    unsigned loff = (unsigned)(unsigned long long)(AS3 void*)lds;
    asm volatile("global_load_async_to_lds_b128 %0, %1, off"
                 :: "v"(loff), "v"(g) : "memory");
#endif
}

__device__ inline void wait_async0() {
#if __has_builtin(__builtin_amdgcn_s_wait_asynccnt)
    __builtin_amdgcn_s_wait_asynccnt(0);
#else
    asm volatile("s_wait_asynccnt 0x0" ::: "memory");
#endif
}

// A fragment (16x32 bf16, M x K) from LDS tile As[rows][32] row-major.
// ISA layout: lanes 0-15 -> M=0..15, g=lane/16 selects K-chunk:
//   element e<8 : k = g*8 + e ; element e>=8 : k = 16 + g*8 + (e-8)
__device__ inline v16bf load_a_frag(const bf16* As, int lane, int row0) {
    int m = row0 + (lane & 15);
    int g = lane >> 4;
    v8bf lo = *(const v8bf*)(As + m * 32 + g * 8);
    v8bf hi = *(const v8bf*)(As + m * 32 + 16 + g * 8);
    v16bf a;
#pragma unroll
    for (int i = 0; i < 8; ++i) { a[i] = lo[i]; a[i + 8] = hi[i]; }
    return a;
}

// B fragment (32x16 bf16, K x N) from LDS tile Bs[32][ldn] row-major.
// ISA layout: lanes 0-15 -> N=0..15, g=lane/16 selects K-chunk:
//   element e<8 : k = g*8 + e ; element e>=8 : k = 16 + g*8 + (e-8)
__device__ inline v16bf load_b_frag(const bf16* Bs, int lane, int col0, int ldn) {
    int n = col0 + (lane & 15);
    int g = lane >> 4;
    v16bf b;
#pragma unroll
    for (int i = 0; i < 8; ++i) {
        b[i] = Bs[(g * 8 + i) * ldn + n];
        b[i + 8] = Bs[(16 + g * 8 + i) * ldn + n];
    }
    return b;
}

// ---------------------------------------------------------------------------
// Kernel 0: convert expert weights fp32 -> bf16
// ---------------------------------------------------------------------------
__global__ __launch_bounds__(256) void cvt_we_kernel(const float* __restrict__ We,
                                                     bf16* __restrict__ We_bf,
                                                     int total) {
    int i = (blockIdx.x * 256 + threadIdx.x) * 4;
    if (i >= total) return;
    float4 f = *(const float4*)(We + i);
    bf16* d = We_bf + i;
    d[0] = (bf16)f.x; d[1] = (bf16)f.y; d[2] = (bf16)f.z; d[3] = (bf16)f.w;
}

// ---------------------------------------------------------------------------
// Kernel 1: gate MLP (WMMA bf16) + softmax + top2; also emits x_bf16.
// Block: 256 threads (8 waves), 64 tokens per block.
// ---------------------------------------------------------------------------
__global__ __launch_bounds__(256) void gate_kernel(
    const float* __restrict__ x, const float* __restrict__ Wg1,
    const float* __restrict__ bg1, const float* __restrict__ Wg2,
    const float* __restrict__ bg2,
    bf16* __restrict__ x_bf, int* __restrict__ top_i, float* __restrict__ top_g,
    int* __restrict__ counts) {

    __shared__ __align__(32) bf16 As[64 * 32];
    __shared__ __align__(32) bf16 Bs[32 * DIM_H];
    __shared__ float hs[64 * DIM_H];
    __shared__ float Wg2s[DIM_H * DIM_E];
    __shared__ float bg1s[DIM_H];
    __shared__ float bg2s[DIM_E];
    __shared__ float lgt[64 * DIM_E];

    const int tid  = threadIdx.x;
    const int lane = tid & 31;
    const int w    = tid >> 5;
    const int rg   = w & 3;   // row group (16 rows)
    const int cg   = w >> 2;  // col group (64 cols)
    const int row0 = blockIdx.x * 64;

    // preload small gate params
#pragma unroll
    for (int q = 0; q < 8; ++q) Wg2s[tid + q * 256] = Wg2[tid + q * 256];
    if (tid < DIM_H) bg1s[tid] = bg1[tid];
    if (tid < DIM_E) bg2s[tid] = bg2[tid];

    v8f acc[4];
#pragma unroll
    for (int s = 0; s < 4; ++s) acc[s] = v8f_zero();

    for (int kb = 0; kb < DIM_D; kb += 32) {
        __syncthreads();
        // stage A (64x32): thread -> row tid/4, 8 consecutive k
        {
            int r = tid >> 2, c = (tid & 3) * 8;
            const float* src = x + (size_t)(row0 + r) * DIM_D + kb + c;
            float4 f0 = *(const float4*)src;
            float4 f1 = *(const float4*)(src + 4);
            v8bf p;
            p[0] = (bf16)f0.x; p[1] = (bf16)f0.y; p[2] = (bf16)f0.z; p[3] = (bf16)f0.w;
            p[4] = (bf16)f1.x; p[5] = (bf16)f1.y; p[6] = (bf16)f1.z; p[7] = (bf16)f1.w;
            *(v8bf*)(As + r * 32 + c) = p;
            *(v8bf*)(x_bf + (size_t)(row0 + r) * DIM_D + kb + c) = p;  // fused convert-out
        }
        // stage B = Wg1[kb..kb+31][0..127]: thread -> row tid/8, 16 cols
        {
            int rr = tid >> 3, cc = (tid & 7) * 16;
            const float* ws = Wg1 + (size_t)(kb + rr) * DIM_H + cc;
            bf16* dst = Bs + rr * DIM_H + cc;
#pragma unroll
            for (int q = 0; q < 4; ++q) {
                float4 f = *(const float4*)(ws + q * 4);
                dst[q * 4 + 0] = (bf16)f.x; dst[q * 4 + 1] = (bf16)f.y;
                dst[q * 4 + 2] = (bf16)f.z; dst[q * 4 + 3] = (bf16)f.w;
            }
        }
        __syncthreads();
        v16bf a = load_a_frag(As, lane, rg * 16);
#pragma unroll
        for (int s = 0; s < 4; ++s) {
            v16bf b = load_b_frag(Bs, lane, cg * 64 + s * 16, DIM_H);
            acc[s] = __builtin_amdgcn_wmma_f32_16x16x32_bf16(
                false, a, false, b, (short)0, acc[s], false, false);
        }
    }
    __syncthreads();

    // h = gelu(acc + bg1) -> LDS
    {
        int g = lane >> 4, nloc = lane & 15;
#pragma unroll
        for (int s = 0; s < 4; ++s) {
#pragma unroll
            for (int v = 0; v < 8; ++v) {
                int m = rg * 16 + v + 8 * g;
                int n = cg * 64 + s * 16 + nloc;
                float val = acc[s][v] + bg1s[n];
                hs[m * DIM_H + n] = 0.5f * val * (1.0f + erff(val * 0.70710678118654752f));
            }
        }
    }
    __syncthreads();

    // logits = h @ Wg2 + bg2 (small K=128, VALU)
    {
        int t = tid & 63;
        int jbase = (tid >> 6) * 4;
#pragma unroll
        for (int jj = 0; jj < 4; ++jj) {
            int j = jbase + jj;
            float s = bg2s[j];
            const float* hrow = hs + t * DIM_H;
#pragma unroll 8
            for (int i = 0; i < DIM_H; ++i) s += hrow[i] * Wg2s[i * DIM_E + j];
            lgt[t * DIM_E + j] = s;
        }
    }
    __syncthreads();

    // softmax + 1e-4, top-2, routing outputs
    if (tid < 64) {
        float p[DIM_E];
        float mx = -1e30f;
#pragma unroll
        for (int j = 0; j < DIM_E; ++j) { p[j] = lgt[tid * DIM_E + j]; mx = fmaxf(mx, p[j]); }
        float sum = 0.0f;
#pragma unroll
        for (int j = 0; j < DIM_E; ++j) { p[j] = __expf(p[j] - mx); sum += p[j]; }
        float inv = 1.0f / sum;
        int i1 = 0, i2 = 0; float v1 = -1.0f, v2 = -1.0f;
#pragma unroll
        for (int j = 0; j < DIM_E; ++j) {
            float pr = p[j] * inv + 1e-4f;
            if (pr > v1)      { v2 = v1; i2 = i1; v1 = pr; i1 = j; }
            else if (pr > v2) { v2 = pr; i2 = j; }
        }
        int n = row0 + tid;
        top_i[n * 2 + 0] = i1; top_i[n * 2 + 1] = i2;
        top_g[n * 2 + 0] = v1; top_g[n * 2 + 1] = v2;
        atomicAdd(&counts[i1], 1);
        atomicAdd(&counts[i2], 1);
    }
}

// ---------------------------------------------------------------------------
// Kernel 2: 16-entry prefix sum -> offsets, cursors
// ---------------------------------------------------------------------------
__global__ void prefix_kernel(const int* __restrict__ counts,
                              int* __restrict__ offs, int* __restrict__ cursors) {
    if (threadIdx.x == 0) {
        int s = 0;
        for (int e = 0; e < DIM_E; ++e) { offs[e] = s; cursors[e] = s; s += counts[e]; }
    }
}

// ---------------------------------------------------------------------------
// Kernel 3: scatter (token, gate) into per-expert buckets
// ---------------------------------------------------------------------------
__global__ __launch_bounds__(256) void bucket_kernel(
    const int* __restrict__ top_i, const float* __restrict__ top_g,
    int* __restrict__ cursors, int* __restrict__ btok, float* __restrict__ bgate,
    int N) {
    int n = blockIdx.x * 256 + threadIdx.x;
    if (n >= N) return;
#pragma unroll
    for (int k = 0; k < 2; ++k) {
        int e = top_i[n * 2 + k];
        int pos = atomicAdd(&cursors[e], 1);
        btok[pos]  = n;
        bgate[pos] = top_g[n * 2 + k];
    }
}

// ---------------------------------------------------------------------------
// Kernel 4: per-expert GEMM (gathered rows), WMMA bf16, async double-buffered
// LDS staging, gated f32 scatter-add.
// Block tile: 128 tokens x 128 outs, K-step 32. 8 waves, each 32x64.
// grid: (4 n-blocks, worst-case m-blocks, 16 experts); block: 256 threads
// ---------------------------------------------------------------------------
__global__ __launch_bounds__(256) void expert_kernel(
    const bf16* __restrict__ x_bf, const bf16* __restrict__ We_bf,
    const float* __restrict__ be,
    const int* __restrict__ offs, const int* __restrict__ counts,
    const int* __restrict__ btok, const float* __restrict__ bgate,
    float* __restrict__ out) {

    const int e   = blockIdx.z;
    const int cnt = counts[e];
    const int mb  = blockIdx.y;
    if (mb * 128 >= cnt) return;
    const int nb  = blockIdx.x;  // 128-wide col block

    __shared__ __align__(32) bf16 As[2][128 * 32];   // 2 x 8KB
    __shared__ __align__(32) bf16 Bs[2][32 * 128];   // 2 x 8KB
    __shared__ int   tokS[128];
    __shared__ float gS[128];
    __shared__ float beS[128];

    const int tid  = threadIdx.x;
    const int lane = tid & 31;
    const int w    = tid >> 5;
    const int rg   = w & 3;    // 32-row group
    const int cg   = w >> 2;   // 64-col group

    const int base = offs[e] + mb * 128;
    if (tid < 128) {
        int slot = mb * 128 + tid;
        if (slot < cnt) { tokS[tid] = btok[base + tid]; gS[tid] = bgate[base + tid]; }
        else            { tokS[tid] = -1;               gS[tid] = 0.0f; }
        beS[tid] = be[e * DIM_O + nb * 128 + tid];
    }
    __syncthreads();

    const bf16* W = We_bf + (size_t)e * DIM_D * DIM_O + nb * 128;

    // Per-thread fixed staging assignments (A: 2 chunks, B: 2 chunks / k-step).
    // A tile 128x32 bf16 = 512 x 16B chunks: chunk q -> row q>>2, col (q&3)*8
    const int ar0 = tid >> 2, ar1 = 64 + (tid >> 2);
    const int ac  = (tid & 3) * 8;
    int at0 = tokS[ar0]; at0 = at0 < 0 ? 0 : at0;
    int at1 = tokS[ar1]; at1 = at1 < 0 ? 0 : at1;
    const bf16* asrc0 = x_bf + (size_t)at0 * DIM_D + ac;
    const bf16* asrc1 = x_bf + (size_t)at1 * DIM_D + ac;
    // B tile 32x128 bf16 = 512 x 16B chunks: chunk q -> row q>>4, col (q&15)*8
    const int br0 = tid >> 4, br1 = 16 + (tid >> 4);
    const int bc  = (tid & 15) * 8;
    const bf16* bsrc0 = W + (size_t)br0 * DIM_O + bc;
    const bf16* bsrc1 = W + (size_t)br1 * DIM_O + bc;

    v8f acc[2][4];
#pragma unroll
    for (int h = 0; h < 2; ++h)
#pragma unroll
        for (int s = 0; s < 4; ++s) acc[h][s] = v8f_zero();

    // prologue: stage k-block 0 into buffer 0
    {
        async_copy_b128(&As[0][ar0 * 32 + ac], asrc0);
        async_copy_b128(&As[0][ar1 * 32 + ac], asrc1);
        async_copy_b128(&Bs[0][br0 * 128 + bc], bsrc0);
        async_copy_b128(&Bs[0][br1 * 128 + bc], bsrc1);
    }
    wait_async0();
    __syncthreads();

    for (int kb = 0; kb < DIM_D; kb += 32) {
        const int cur = (kb >> 5) & 1;
        const int nxt = cur ^ 1;
        if (kb + 32 < DIM_D) {  // stage next k-block while computing
            int kn = kb + 32;
            async_copy_b128(&As[nxt][ar0 * 32 + ac], asrc0 + kn);
            async_copy_b128(&As[nxt][ar1 * 32 + ac], asrc1 + kn);
            async_copy_b128(&Bs[nxt][br0 * 128 + bc], bsrc0 + (size_t)kn * DIM_O);
            async_copy_b128(&Bs[nxt][br1 * 128 + bc], bsrc1 + (size_t)kn * DIM_O);
        }
        v16bf a0 = load_a_frag(&As[cur][0], lane, rg * 32);
        v16bf a1 = load_a_frag(&As[cur][0], lane, rg * 32 + 16);
#pragma unroll
        for (int s = 0; s < 4; ++s) {
            v16bf b = load_b_frag(&Bs[cur][0], lane, cg * 64 + s * 16, 128);
            acc[0][s] = __builtin_amdgcn_wmma_f32_16x16x32_bf16(
                false, a0, false, b, (short)0, acc[0][s], false, false);
            acc[1][s] = __builtin_amdgcn_wmma_f32_16x16x32_bf16(
                false, a1, false, b, (short)0, acc[1][s], false, false);
        }
        wait_async0();
        __syncthreads();
    }

    // writeback: out[tok] += gate * (acc + be)
    {
        int g = lane >> 4, nloc = lane & 15;
#pragma unroll
        for (int h = 0; h < 2; ++h) {
#pragma unroll
            for (int s = 0; s < 4; ++s) {
#pragma unroll
                for (int v = 0; v < 8; ++v) {
                    int m = rg * 32 + h * 16 + v + 8 * g;
                    int tok = tokS[m];
                    if (tok >= 0) {
                        int n = cg * 64 + s * 16 + nloc;
                        float val = (acc[h][s][v] + beS[n]) * gS[m];
                        atomicAdd(out + (size_t)tok * DIM_O + nb * 128 + n, val);
                    }
                }
            }
        }
    }
}

// ---------------------------------------------------------------------------
// Host-side launcher
// ---------------------------------------------------------------------------
extern "C" void kernel_launch(void* const* d_in, const int* in_sizes, int n_in,
                              void* d_out, int out_size, void* d_ws, size_t ws_size,
                              hipStream_t stream) {
    const float* x   = (const float*)d_in[0];
    const float* Wg1 = (const float*)d_in[1];
    const float* bg1 = (const float*)d_in[2];
    const float* Wg2 = (const float*)d_in[3];
    const float* bg2 = (const float*)d_in[4];
    const float* We  = (const float*)d_in[5];
    const float* be  = (const float*)d_in[6];

    const size_t N = (size_t)in_sizes[0] / DIM_D;

    // workspace carve-up
    char* w = (char*)d_ws;
    bf16*  x_bf  = (bf16*)w;                 size_t off = N * DIM_D * sizeof(bf16);
    bf16*  We_bf = (bf16*)(w + off);         off += (size_t)DIM_E * DIM_D * DIM_O * sizeof(bf16);
    int*   top_i = (int*)(w + off);          off += N * 2 * sizeof(int);
    float* top_g = (float*)(w + off);        off += N * 2 * sizeof(float);
    int*   btok  = (int*)(w + off);          off += N * 2 * sizeof(int);
    float* bgate = (float*)(w + off);        off += N * 2 * sizeof(float);
    int*   counts   = (int*)(w + off);       // 16
    int*   offs     = counts + 16;           // 16
    int*   cursors  = counts + 32;           // 16

    (void)hipMemsetAsync(counts, 0, 64, stream);

    // 0: convert expert weights
    {
        int total = DIM_E * DIM_D * DIM_O;
        cvt_we_kernel<<<total / (256 * 4), 256, 0, stream>>>(We, We_bf, total);
    }
    // 1: gate + x_bf16
    gate_kernel<<<(int)(N / 64), 256, 0, stream>>>(x, Wg1, bg1, Wg2, bg2,
                                                   x_bf, top_i, top_g, counts);
    // 2: prefix
    prefix_kernel<<<1, 32, 0, stream>>>(counts, offs, cursors);
    // 3: bucket
    bucket_kernel<<<(int)((N + 255) / 256), 256, 0, stream>>>(top_i, top_g, cursors,
                                                              btok, bgate, (int)N);
    // 4: zero output, then gated expert GEMM scatter-add
    (void)hipMemsetAsync(d_out, 0, (size_t)out_size * sizeof(float), stream);
    dim3 grid(DIM_O / 128, (unsigned)((N + 127) / 128), DIM_E);
    expert_kernel<<<grid, 256, 0, stream>>>(x_bf, We_bf, be, offs, counts,
                                            btok, bgate, (float*)d_out);
}
